// GIN_32332513804324
// MI455X (gfx1250) — matrix operand order repaired and checked by reading, not compiled, not run on hardware
//
#include <hip/hip_runtime.h>

// ---------------------------------------------------------------------------
// GIN forward for MI455X (gfx1250, wave32, WMMA).
// Scatter-bound workload (~1.6 GB of edge traffic @ 23.3 TB/s); GEMMs run on
// v_wmma_f32_16x16x32_f16 (f32 accumulate). Layer kernel fuses
// (h+agg) -> relu(GEMM1) -> relu(GEMM2) -> +skip.
// Round 3: readfirstlane on the wave row base so tile-bound checks lower to
// scalar branches (s_cbranch_scc) instead of exec-mask predication.
// ---------------------------------------------------------------------------

#define NN    100000   // nodes
#define NE    1000000  // edges
#define NG    128      // graphs
#define IND   32       // input dim
#define HID   64       // hidden dim
#define LAT   32       // latent dim

typedef __attribute__((ext_vector_type(16))) _Float16 v16h;
typedef __attribute__((ext_vector_type(8)))  float    v8f;

// A-fragment (16x32 f16 tile, ISA 7.12.2 dense 16-bit layout):
// lane m = lane&15 holds row M=m; lanes>=16 add +8 to K.
// VGPR v<4 : K pair base 2v ; v>=4 : 2v+8.
__device__ __forceinline__ v16h load_a_frag(const _Float16* lds, int rowBase,
                                            int stride, int kBase, int m, int hi) {
  const _Float16* p = lds + (rowBase + m) * stride + kBase + (hi ? 8 : 0);
  v16h a;
#pragma unroll
  for (int v = 0; v < 8; ++v) {
    const int k0 = (v < 4) ? (2 * v) : (2 * v + 8);
    a[2 * v]     = p[k0];
    a[2 * v + 1] = p[k0 + 1];
  }
  return a;
}

// B-fragment: weights staged transposed [N][K] f16; per lane (col n = lane&15)
// the fragment is K = kBase + (hi?8:0) + {0..7, 16..23} (same order as A).
__device__ __forceinline__ v16h load_b_frag(const _Float16* ldsWT, int n,
                                            int kStride, int kBase, int hi) {
  const _Float16* p = ldsWT + n * kStride + kBase + (hi ? 8 : 0);
  v16h b;
#pragma unroll
  for (int v = 0; v < 16; ++v) b[v] = p[v < 8 ? v : v + 8];
  return b;
}

// ---------------------------------------------------------------------------
// h = x @ w_in + b_in    [NN,32] x [32,64] ; K=32 => one WMMA per 16x16 tile
// ---------------------------------------------------------------------------
#define XSTR 40  // padded LDS row stride (breaks bank periodicity)
__global__ void __launch_bounds__(256) gin_input_proj(
    const float* __restrict__ x, const float* __restrict__ w_in,
    const float* __restrict__ b_in, float* __restrict__ h) {
  __shared__ __align__(32) _Float16 ldsW[HID * IND];    // [n][k] transposed
  __shared__ __align__(32) _Float16 ldsX[128 * XSTR];

  const int tid  = threadIdx.x;
  const int lane = tid & 31;
  const int wave = tid >> 5;
  const int m    = lane & 15;
  const int hi   = lane >> 4;
  const int rowBlock = blockIdx.x * 128;

  for (int i = tid; i < HID * IND; i += 256) {
    const int n = i >> 5, k = i & 31;
    ldsW[i] = (_Float16)w_in[k * HID + n];
  }
  for (int i = tid; i < 128 * IND; i += 256) {
    const int r = i >> 5, k = i & 31;
    const int row = rowBlock + r;
    ldsX[r * XSTR + k] = (_Float16)(row < NN ? x[row * IND + k] : 0.0f);
  }
  __syncthreads();

  const int waveRow  = wave * 16;
  // wave-uniform scalar: bound checks lower to s_cmp/s_cbranch, not exec masks
  const int waveBase = __builtin_amdgcn_readfirstlane(rowBlock + waveRow);
  if (waveBase >= NN) return;                 // scalar branch, past barrier
  const bool fullTile = (waveBase + 16 <= NN);

  const v16h a = load_a_frag(ldsX, waveRow, XSTR, 0, m, hi);

#pragma unroll
  for (int t = 0; t < 4; ++t) {
    const int n = t * 16 + m;
    const v16h b = load_b_frag(ldsW, n, IND, 0, hi);
    const float bias = b_in[n];
    v8f c = {bias, bias, bias, bias, bias, bias, bias, bias};
    c = __builtin_amdgcn_wmma_f32_16x16x32_f16(false, a, false, b,
                                               (short)0, c, false, false);
    if (fullTile) {
#pragma unroll
      for (int r = 0; r < 8; ++r) {
        const int row = waveBase + (hi ? 8 + r : r);
        h[row * HID + n] = c[r];
      }
    } else {
#pragma unroll
      for (int r = 0; r < 8; ++r) {
        const int row = waveBase + (hi ? 8 + r : r);
        if (row < NN) h[row * HID + n] = c[r];
      }
    }
  }
}

// ---------------------------------------------------------------------------
// agg[dst] += h[src] over all edges. 16 lanes/edge, float4 + f32 atomics.
// ---------------------------------------------------------------------------
__global__ void __launch_bounds__(256) gin_aggregate(
    const long long* __restrict__ ei, const float* __restrict__ h,
    float* __restrict__ agg) {
  const int idx = blockIdx.x * 256 + threadIdx.x;
  const int e  = idx >> 4;
  const int c4 = (idx & 15) << 2;
  if (e >= NE) return;
  const int src = (int)ei[e];
  const int dst = (int)ei[NE + e];
  const float4 v = *(const float4*)(h + src * HID + c4);
  float* p = agg + dst * HID + c4;
  unsafeAtomicAdd(p + 0, v.x);
  unsafeAtomicAdd(p + 1, v.y);
  unsafeAtomicAdd(p + 2, v.z);
  unsafeAtomicAdd(p + 3, v.w);
}

// ---------------------------------------------------------------------------
// Fused GIN layer: m = h+agg ; m = relu(m@w1+b1) ; m = relu(m@w2+b2) ; h=m+skip
// 256 threads = 8 waves, 128 rows per block. K=64 => 2 WMMAs per tile.
// ---------------------------------------------------------------------------
#define MSTR 72  // padded LDS row stride for activation tiles
__global__ void __launch_bounds__(256) gin_layer(
    const float* __restrict__ h_in, const float* __restrict__ agg,
    const float* __restrict__ w1, const float* __restrict__ b1,
    const float* __restrict__ w2, const float* __restrict__ b2,
    float* __restrict__ h_out) {
  __shared__ __align__(32) _Float16 ldsW1[HID * HID];   // transposed [n][k]
  __shared__ __align__(32) _Float16 ldsW2[HID * HID];
  __shared__ __align__(32) _Float16 ldsM[128 * MSTR];
  __shared__ __align__(32) _Float16 ldsT[128 * MSTR];

  const int tid  = threadIdx.x;
  const int lane = tid & 31;
  const int wave = tid >> 5;
  const int m    = lane & 15;
  const int hi   = lane >> 4;
  const int rowBlock = blockIdx.x * 128;

  for (int i = tid; i < HID * HID; i += 256) {
    const int n = i >> 6, k = i & 63;
    ldsW1[i] = (_Float16)w1[k * HID + n];
    ldsW2[i] = (_Float16)w2[k * HID + n];
  }
  for (int i = tid; i < 128 * HID; i += 256) {
    const int r = i >> 6, c = i & 63;
    const int row = rowBlock + r;
    float v = 0.0f;
    if (row < NN) v = h_in[row * HID + c] + agg[row * HID + c];
    ldsM[r * MSTR + c] = (_Float16)v;
  }
  __syncthreads();

  const int waveRow  = wave * 16;
  // wave-uniform scalar: bound checks lower to s_cmp/s_cbranch, not exec masks
  const int waveBase = __builtin_amdgcn_readfirstlane(rowBlock + waveRow);
  if (waveBase >= NN) return;                 // scalar branch, past barrier
  const bool fullTile = (waveBase + 16 <= NN);

  // ---- GEMM1: relu(m @ w1 + b1) -> ldsT (wave-local rows, DS in-order) ----
  {
    const v16h a0 = load_a_frag(ldsM, waveRow, MSTR, 0,  m, hi);
    const v16h a1 = load_a_frag(ldsM, waveRow, MSTR, 32, m, hi);
#pragma unroll
    for (int t = 0; t < 4; ++t) {
      const int n = t * 16 + m;
      const v16h b0  = load_b_frag(ldsW1, n, HID, 0,  hi);
      const v16h b1f = load_b_frag(ldsW1, n, HID, 32, hi);
      const float bias = b1[n];
      v8f c = {bias, bias, bias, bias, bias, bias, bias, bias};
      c = __builtin_amdgcn_wmma_f32_16x16x32_f16(false, a0, false, b0,
                                                 (short)0, c, false, false);
      c = __builtin_amdgcn_wmma_f32_16x16x32_f16(false, a1, false, b1f,
                                                 (short)0, c, false, false);
#pragma unroll
      for (int r = 0; r < 8; ++r) {
        const int rr = waveRow + (hi ? 8 + r : r);
        ldsT[rr * MSTR + n] = (_Float16)fmaxf(c[r], 0.0f);
      }
    }
  }

  // ---- GEMM2: relu(t @ w2 + b2) + skip -> h_out ----
  {
    const v16h a0 = load_a_frag(ldsT, waveRow, MSTR, 0,  m, hi);
    const v16h a1 = load_a_frag(ldsT, waveRow, MSTR, 32, m, hi);
#pragma unroll
    for (int t = 0; t < 4; ++t) {
      const int n = t * 16 + m;
      const v16h b0  = load_b_frag(ldsW2, n, HID, 0,  hi);
      const v16h b1f = load_b_frag(ldsW2, n, HID, 32, hi);
      const float bias = b2[n];
      v8f c = {bias, bias, bias, bias, bias, bias, bias, bias};
      c = __builtin_amdgcn_wmma_f32_16x16x32_f16(false, a0, false, b0,
                                                 (short)0, c, false, false);
      c = __builtin_amdgcn_wmma_f32_16x16x32_f16(false, a1, false, b1f,
                                                 (short)0, c, false, false);
      if (fullTile) {
#pragma unroll
        for (int r = 0; r < 8; ++r) {
          const int row = waveBase + (hi ? 8 + r : r);
          h_out[row * HID + n] = fmaxf(c[r], 0.0f) + h_in[row * HID + n];
        }
      } else {
#pragma unroll
        for (int r = 0; r < 8; ++r) {
          const int row = waveBase + (hi ? 8 + r : r);
          if (row < NN)
            h_out[row * HID + n] = fmaxf(c[r], 0.0f) + h_in[row * HID + n];
        }
      }
    }
  }
}

// ---------------------------------------------------------------------------
// pooled[batch[node]] += h[node]  (batch sorted, 16 lanes/node)
// ---------------------------------------------------------------------------
__global__ void __launch_bounds__(256) gin_pool(
    const long long* __restrict__ batch, const float* __restrict__ h,
    float* __restrict__ pooled) {
  const int idx = blockIdx.x * 256 + threadIdx.x;
  const int node = idx >> 4;
  const int c4   = (idx & 15) << 2;
  if (node >= NN) return;
  const int g = (int)batch[node];
  const float4 v = *(const float4*)(h + node * HID + c4);
  float* p = pooled + g * HID + c4;
  unsafeAtomicAdd(p + 0, v.x);
  unsafeAtomicAdd(p + 1, v.y);
  unsafeAtomicAdd(p + 2, v.z);
  unsafeAtomicAdd(p + 3, v.w);
}

// out = pooled @ w_fc + b_fc   [128,64] x [64,32]  (tiny)
__global__ void __launch_bounds__(256) gin_head(
    const float* __restrict__ pooled, const float* __restrict__ w_fc,
    const float* __restrict__ b_fc, float* __restrict__ out) {
  const int idx = blockIdx.x * 256 + threadIdx.x;
  if (idx >= NG * LAT) return;
  const int g = idx >> 5, j = idx & 31;
  float s = b_fc[j];
#pragma unroll
  for (int k = 0; k < HID; ++k) s += pooled[g * HID + k] * w_fc[k * LAT + j];
  out[idx] = s;
}

__global__ void __launch_bounds__(256) gin_zero4(float4* __restrict__ p, int n4) {
  const int i = blockIdx.x * 256 + threadIdx.x;
  if (i < n4) p[i] = make_float4(0.f, 0.f, 0.f, 0.f);
}

// ---------------------------------------------------------------------------
extern "C" void kernel_launch(void* const* d_in, const int* in_sizes, int n_in,
                              void* d_out, int out_size, void* d_ws, size_t ws_size,
                              hipStream_t stream) {
  (void)in_sizes; (void)n_in; (void)out_size; (void)ws_size;
  const float*     x     = (const float*)d_in[0];
  const long long* ei    = (const long long*)d_in[1];   // int64 [2, NE]
  const long long* batch = (const long long*)d_in[2];   // int64 [NN]
  const float*     w_in  = (const float*)d_in[3];
  const float*     b_in  = (const float*)d_in[4];
  const float*     w1    = (const float*)d_in[5];       // [3,64,64]
  const float*     b1    = (const float*)d_in[6];       // [3,64]
  const float*     w2    = (const float*)d_in[7];
  const float*     b2    = (const float*)d_in[8];
  const float*     w_fc  = (const float*)d_in[9];
  const float*     b_fc  = (const float*)d_in[10];
  float*           out   = (float*)d_out;

  // workspace layout (f32): hA | hB | agg | pooled  (~77 MB)
  const size_t NH = (size_t)NN * HID;
  float* hA     = (float*)d_ws;
  float* hB     = hA + NH;
  float* agg    = hB + NH;
  float* pooled = agg + NH;

  const int rowBlocks = (NN + 127) / 128;               // 782
  const int aggN4     = (int)(NH / 4);                  // 1.6M float4

  gin_input_proj<<<rowBlocks, 256, 0, stream>>>(x, w_in, b_in, hA);

  float* cur = hA;
  float* nxt = hB;
  for (int l = 0; l < 3; ++l) {
    gin_zero4<<<(aggN4 + 255) / 256, 256, 0, stream>>>((float4*)agg, aggN4);
    gin_aggregate<<<(NE * 16) / 256, 256, 0, stream>>>(ei, cur, agg);
    gin_layer<<<rowBlocks, 256, 0, stream>>>(cur, agg,
                                             w1 + (size_t)l * HID * HID, b1 + l * HID,
                                             w2 + (size_t)l * HID * HID, b2 + l * HID,
                                             nxt);
    float* t = cur; cur = nxt; nxt = t;
  }

  gin_zero4<<<(NG * HID / 4 + 255) / 256, 256, 0, stream>>>((float4*)pooled,
                                                            NG * HID / 4);
  gin_pool<<<(NN * 16 + 255) / 256, 256, 0, stream>>>(batch, cur, pooled);
  gin_head<<<(NG * LAT + 255) / 256, 256, 0, stream>>>(pooled, w_fc, b_fc, out);
}
